// FlexBertUnpadRopeAttention_43731357008513
// MI455X (gfx1250) — compile-verified, matrix operand order checked by reading, not run on hardware
//
#include <hip/hip_runtime.h>
#include <hip/hip_bf16.h>
#include <stdint.h>

typedef __bf16 bf16_t;
typedef __attribute__((ext_vector_type(16))) __bf16 v16bf;
typedef __attribute__((ext_vector_type(8)))  __bf16 v8bf;
typedef __attribute__((ext_vector_type(8)))  float  v8f;
typedef __attribute__((ext_vector_type(4)))  unsigned int v4u;
typedef __attribute__((ext_vector_type(4)))  int v4i;
typedef __attribute__((ext_vector_type(8)))  int v8i;

#define H_      16
#define HD_     64
#define DIM_    1024
#define B_      4
#define MAXLEN_ 2048
#define TOTAL_  5120
#define QKVLD_  3072

// ---------------------------------------------------------------------------
// 16x32 bf16 fragment loader (A-layout per CDNA5 ISA): lanes 0-15 hold rows
// M=0..15 with K chunks {0..7,16..23}; lanes 16-31 same rows, K chunks
// {8..15,24..31}. Works for global (global_load_b128) and LDS-derived
// pointers (ds_load_b128 via addrspace inference). B fragments for NT GEMMs
// use the same pattern with row0 = column-tile base.
// ---------------------------------------------------------------------------
__device__ __forceinline__ v16bf load_frag_g(const bf16_t* __restrict__ base,
                                             int row0, int k0, int ld) {
  const int lane = threadIdx.x & 31;
  const int r    = row0 + (lane & 15);
  const int kb   = k0 + ((lane & 16) ? 8 : 0);
  const bf16_t* p = base + (size_t)r * (size_t)ld + kb;
  union { v16bf v; v8bf h[2]; } u;
  u.h[0] = *(const v8bf*)(p);
  u.h[1] = *(const v8bf*)(p + 16);
  return u.v;
}

__device__ __forceinline__ v8f wmma_bf16(v16bf a, v16bf b, v8f c) {
  return __builtin_amdgcn_wmma_f32_16x16x32_bf16(false, a, false, b,
                                                 (short)0, c, false, false);
}

// ---------------------------------------------------------------------------
// Tensor Data Mover: async 2D tile load Global -> LDS (TENSORcnt tracked).
// D# built per CDNA5 ISA ch.8: data_size=8B units, count=1, type=2 ("image"),
// 2D tile (tile_dim2..4 = 0 => unused, groups 2/3 zero).
//   lds_off    : LDS byte offset of destination (contiguous tile)
//   gaddr      : global address of tile start
//   tile0_u8   : tile width in 8-byte units   (== tensor_dim0)
//   tile1      : tile height in rows          (== tensor_dim1)
//   stride0_u8 : row stride in 8-byte units
// ---------------------------------------------------------------------------
__device__ __forceinline__ void tdm_load_2d(unsigned lds_off, const void* gaddr,
                                            unsigned tile0_u8, unsigned tile1,
                                            unsigned stride0_u8) {
  const unsigned long long ga = (unsigned long long)(size_t)gaddr;
  v4u g0 = { 1u,                                   // count=1 (valid descriptor)
             lds_off,                              // lds_addr
             (unsigned)(ga & 0xffffffffu),         // global_addr[31:0]
             (unsigned)((ga >> 32) & 0x01ffffffu)  // global_addr[56:32]
                 | (2u << 30) };                   // type=2
  v8i g1 = { (int)0x00030000u,                     // data_size=3 (8B)
             (int)(tile0_u8 << 16),                // tensor_dim0[15:0] @63:48
             (int)(tile1 << 16),                   // tensor_dim1[15:0] @95:80
             (int)(tile0_u8 << 16),                // tile_dim0 @127:112
             (int)tile1,                           // tile_dim1 @143:128
             (int)stride0_u8,                      // tensor_dim0_stride[31:0]
             0, 0 };
  v4i z4 = { 0, 0, 0, 0 };
#if __clang_major__ >= 23
  v8i z8 = { 0, 0, 0, 0, 0, 0, 0, 0 };
  __builtin_amdgcn_tensor_load_to_lds(g0, g1, z4, z4, z8, 0);
#else
  __builtin_amdgcn_tensor_load_to_lds(g0, g1, z4, z4, 0);
#endif
}

// ---------------------------------------------------------------------------
// fp32 -> bf16 conversion, 8 elements / thread
// ---------------------------------------------------------------------------
__global__ void cvt_kernel(const float* __restrict__ in,
                           bf16_t* __restrict__ out, int n8) {
  int i = blockIdx.x * blockDim.x + threadIdx.x;
  if (i >= n8) return;
  const float4* p = (const float4*)in + (size_t)i * 2;
  float4 x = p[0];
  float4 y = p[1];
  v8bf o = { (__bf16)x.x, (__bf16)x.y, (__bf16)x.z, (__bf16)x.w,
             (__bf16)y.x, (__bf16)y.y, (__bf16)y.z, (__bf16)y.w };
  ((v8bf*)out)[i] = o;
}

// ---------------------------------------------------------------------------
// QKV GEMM: qkv[5120,3072] = hs @ Wqkv^T. One wave per 16x64 tile,
// register double-buffered over K so WMMAs overlap next-k loads.
// ---------------------------------------------------------------------------
__global__ void __launch_bounds__(256)
qkv_gemm(const bf16_t* __restrict__ hs, const bf16_t* __restrict__ w,
         bf16_t* __restrict__ out) {
  const int wave = (blockIdx.x * blockDim.x + threadIdx.x) >> 5;
  const int NT = QKVLD_ / 64;              // 48 column tiles
  const int mt = wave / NT;
  const int nt = wave % NT;
  if (mt >= TOTAL_ / 16) return;
  const int row0 = mt * 16, col0 = nt * 64;

  v8f acc[4] = {};
  v16bf a  = load_frag_g(hs, row0, 0, DIM_);
  v16bf b0 = load_frag_g(w, col0 +  0, 0, DIM_);
  v16bf b1 = load_frag_g(w, col0 + 16, 0, DIM_);
  v16bf b2 = load_frag_g(w, col0 + 32, 0, DIM_);
  v16bf b3 = load_frag_g(w, col0 + 48, 0, DIM_);
  for (int k0 = 0; k0 < DIM_; k0 += 32) {
    v16bf an = a, c0 = b0, c1 = b1, c2 = b2, c3 = b3;
    const int kn = k0 + 32;
    if (kn < DIM_) {
      an = load_frag_g(hs, row0, kn, DIM_);
      c0 = load_frag_g(w, col0 +  0, kn, DIM_);
      c1 = load_frag_g(w, col0 + 16, kn, DIM_);
      c2 = load_frag_g(w, col0 + 32, kn, DIM_);
      c3 = load_frag_g(w, col0 + 48, kn, DIM_);
    }
    acc[0] = wmma_bf16(a, b0, acc[0]);
    acc[1] = wmma_bf16(a, b1, acc[1]);
    acc[2] = wmma_bf16(a, b2, acc[2]);
    acc[3] = wmma_bf16(a, b3, acc[3]);
    a = an; b0 = c0; b1 = c1; b2 = c2; b3 = c3;
  }
  const int lane = threadIdx.x & 31;
  const int rofs = (lane & 16) ? 8 : 0;
  const int col  = lane & 15;
#pragma unroll
  for (int j = 0; j < 4; ++j)
#pragma unroll
    for (int r = 0; r < 8; ++r)
      out[(size_t)(row0 + r + rofs) * QKVLD_ + col0 + j * 16 + col] =
          (__bf16)acc[j][r];
}

// ---------------------------------------------------------------------------
// RoPE + scatter:
//   Qp, Kp : [B, MAXLEN, H, HD] bf16  (Q pre-scaled by 1/sqrt(HD))
//   Vt     : [B, H, HD, MAXLEN] bf16  (transposed: PV B-frag K-contiguous)
// Padded regions are never read (phantom keys handled analytically).
// ---------------------------------------------------------------------------
__global__ void __launch_bounds__(256)
rope_scatter(const bf16_t* __restrict__ qkv, const long long* __restrict__ idx,
             bf16_t* __restrict__ Qp, bf16_t* __restrict__ Kp,
             bf16_t* __restrict__ Vt) {
  const int t = blockIdx.x;
  const long long ix = idx[t];
  const int b   = (int)(ix >> 11);
  const int pos = (int)(ix & 2047);
  const bf16_t* row = qkv + (size_t)t * QKVLD_;

  for (int i = threadIdx.x; i < DIM_; i += 256) {
    const int h = i >> 6, d = i & 63;
    Vt[(((size_t)b * H_ + h) * HD_ + d) * MAXLEN_ + pos] = row[2 * DIM_ + i];
  }
  const float scale = 0.125f;
  for (int i = threadIdx.x; i < 512; i += 256) {
    const int h = i >> 5, j = i & 31;
    const float invf = __powf(10000.0f, -(float)(2 * j) * (1.0f / 64.0f));
    const float fr = (float)pos * invf;
    float c, s;
    __sincosf(fr, &s, &c);
    const size_t ob = ((size_t)b * MAXLEN_ + pos) * DIM_ + h * HD_;
    float x1 = (float)row[h * HD_ + j];
    float x2 = (float)row[h * HD_ + j + 32];
    Qp[ob + j]      = (__bf16)((x1 * c - x2 * s) * scale);
    Qp[ob + j + 32] = (__bf16)((x2 * c + x1 * s) * scale);
    x1 = (float)row[DIM_ + h * HD_ + j];
    x2 = (float)row[DIM_ + h * HD_ + j + 32];
    Kp[ob + j]      = (__bf16)(x1 * c - x2 * s);
    Kp[ob + j + 32] = (__bf16)(x2 * c + x1 * s);
  }
}

// ---------------------------------------------------------------------------
// Flash attention. Block = 4 waves = 64 consecutive queries of one (b,h).
// K/V 32-key tiles are TDM-DMA'd into double-buffered LDS (shared by all 4
// waves: 4x less L2 traffic), wave0 drives the DMA + s_wait_tensorcnt,
// one workgroup barrier per key block. Phantom padded keys (score 0,
// value 0) are folded in analytically.
// ---------------------------------------------------------------------------
__global__ void __launch_bounds__(128)
attn_kernel(const bf16_t* __restrict__ Qp, const bf16_t* __restrict__ Kp,
            const bf16_t* __restrict__ Vt, const int* __restrict__ cu,
            bf16_t* __restrict__ out) {
  __shared__ alignas(16) bf16_t ldsK[2][32 * 64];   // keys x d
  __shared__ alignas(16) bf16_t ldsV[2][64 * 32];   // d x pos (from Vt)
  __shared__ alignas(16) bf16_t ldsP[4][16 * 32];   // per-wave P transpose
  const int wib  = threadIdx.x >> 5;
  const int lane = threadIdx.x & 31;
  const int blk  = blockIdx.x;

  // decode (b, h, 64-query block) from flat block id via cu_seqlens
  int b = 0, base = 0, L = 0;
  for (; b < B_; ++b) {
    L = cu[b + 1] - cu[b];
    const int nb = (L / 64) * H_;
    if (blk < base + nb) break;
    base += nb;
  }
  if (b >= B_) return;
  const int rem = blk - base;
  const int nqb = L / 64;
  const int h   = rem / nqb;
  const int qt  = (rem % nqb) * 4 + wib;       // this wave's 16-query tile

  const bf16_t* qb  = Qp + (size_t)b * MAXLEN_ * DIM_ + h * HD_;
  const bf16_t* kbp = Kp + (size_t)b * MAXLEN_ * DIM_ + h * HD_;
  const bf16_t* vbp = Vt + ((size_t)b * H_ + h) * HD_ * MAXLEN_;

  const v16bf a0 = load_frag_g(qb, qt * 16, 0, DIM_);
  const v16bf a1 = load_frag_g(qb, qt * 16, 32, DIM_);

  v8f o[4] = {};
  float m[8], l[8];
#pragma unroll
  for (int r = 0; r < 8; ++r) { m[r] = -INFINITY; l[r] = 0.0f; }

  const int rofs = (lane & 16) ? 8 : 0;
  const int col  = lane & 15;

  // prologue: DMA key-block 0 into buffer 0
  if (wib == 0) {
    tdm_load_2d((unsigned)(size_t)&ldsK[0][0], kbp, 16, 32, 256);
    tdm_load_2d((unsigned)(size_t)&ldsV[0][0], vbp, 8, 64, 512);
  }

  int bufc = 0;
  for (int kblk = 0; kblk < L; kblk += 32) {
    if (wib == 0) __builtin_amdgcn_s_wait_tensorcnt(0);  // current tiles landed
    __syncthreads();  // tiles visible; prior reads of the other buffer done
    if (wib == 0 && kblk + 32 < L) {                     // DMA next block
      tdm_load_2d((unsigned)(size_t)&ldsK[bufc ^ 1][0],
                  kbp + (size_t)(kblk + 32) * DIM_, 16, 32, 256);
      tdm_load_2d((unsigned)(size_t)&ldsV[bufc ^ 1][0],
                  vbp + (kblk + 32), 8, 64, 512);
    }

    const bf16_t* kt = &ldsK[bufc][0];
    const bf16_t* vt = &ldsV[bufc][0];

    // S = Q K^T (pre-scaled), two 16x16 subtiles over K=64, from LDS
    v8f s0 = {}, s1 = {};
    s0 = wmma_bf16(a0, load_frag_g(kt, 0, 0, 64), s0);
    s0 = wmma_bf16(a1, load_frag_g(kt, 0, 32, 64), s0);
    s1 = wmma_bf16(a0, load_frag_g(kt, 16, 0, 64), s1);
    s1 = wmma_bf16(a1, load_frag_g(kt, 16, 32, 64), s1);

    // online softmax per row (16 lanes of a half hold one row)
#pragma unroll
    for (int r = 0; r < 8; ++r) {
      float vmax = fmaxf(s0[r], s1[r]);
#pragma unroll
      for (int mk = 8; mk >= 1; mk >>= 1)
        vmax = fmaxf(vmax, __shfl_xor(vmax, mk, 32));
      const float mnew = fmaxf(m[r], vmax);
      const float corr = __expf(m[r] - mnew);
      const float p0 = __expf(s0[r] - mnew);
      const float p1 = __expf(s1[r] - mnew);
      float rs = p0 + p1;
#pragma unroll
      for (int mk = 8; mk >= 1; mk >>= 1)
        rs += __shfl_xor(rs, mk, 32);
      l[r] = l[r] * corr + rs;
      m[r] = mnew;
      o[0][r] *= corr; o[1][r] *= corr; o[2][r] *= corr; o[3][r] *= corr;
      const int row = r + rofs;
      ldsP[wib][row * 32 + col]      = (__bf16)p0;   // C-layout -> LDS
      ldsP[wib][row * 32 + 16 + col] = (__bf16)p1;
    }

    // reload P in A-fragment layout (per-wave LDS; DS ops in-order per wave)
    const bf16_t* lp = &ldsP[wib][(lane & 15) * 32 + ((lane & 16) ? 8 : 0)];
    union { v16bf v; v8bf hh[2]; } up;
    up.hh[0] = *(const v8bf*)(lp);
    up.hh[1] = *(const v8bf*)(lp + 16);

    // O += P V  (V tile is d x pos, so B-frag columns are K-contiguous)
#pragma unroll
    for (int n = 0; n < 4; ++n)
      o[n] = wmma_bf16(up.v, load_frag_g(vt, n * 16, 0, 32), o[n]);

    bufc ^= 1;
  }

  // phantom padded keys: score exactly 0, value 0 -> denominator only
  const float nphantom = (float)(MAXLEN_ - L);
  const int tok0 = cu[b] + qt * 16;
#pragma unroll
  for (int r = 0; r < 8; ++r) {
    const float mf = fmaxf(m[r], 0.0f);
    const float f  = __expf(m[r] - mf);
    const float Z  = l[r] * f + nphantom * __expf(-mf);
    const float g  = f / Z;
#pragma unroll
    for (int n = 0; n < 4; ++n)
      out[(size_t)(tok0 + r + rofs) * DIM_ + h * HD_ + n * 16 + col] =
          (__bf16)(o[n][r] * g);
  }
}

// ---------------------------------------------------------------------------
// Output projection: out[5120,1024] f32 = attn @ Wo^T (double-buffered)
// ---------------------------------------------------------------------------
__global__ void __launch_bounds__(256)
out_gemm(const bf16_t* __restrict__ attn, const bf16_t* __restrict__ w,
         float* __restrict__ out) {
  const int wave = (blockIdx.x * blockDim.x + threadIdx.x) >> 5;
  const int NT = DIM_ / 64;                // 16 column tiles
  const int mt = wave / NT;
  const int nt = wave % NT;
  if (mt >= TOTAL_ / 16) return;
  const int row0 = mt * 16, col0 = nt * 64;

  v8f acc[4] = {};
  v16bf a  = load_frag_g(attn, row0, 0, DIM_);
  v16bf b0 = load_frag_g(w, col0 +  0, 0, DIM_);
  v16bf b1 = load_frag_g(w, col0 + 16, 0, DIM_);
  v16bf b2 = load_frag_g(w, col0 + 32, 0, DIM_);
  v16bf b3 = load_frag_g(w, col0 + 48, 0, DIM_);
  for (int k0 = 0; k0 < DIM_; k0 += 32) {
    v16bf an = a, c0 = b0, c1 = b1, c2 = b2, c3 = b3;
    const int kn = k0 + 32;
    if (kn < DIM_) {
      an = load_frag_g(attn, row0, kn, DIM_);
      c0 = load_frag_g(w, col0 +  0, kn, DIM_);
      c1 = load_frag_g(w, col0 + 16, kn, DIM_);
      c2 = load_frag_g(w, col0 + 32, kn, DIM_);
      c3 = load_frag_g(w, col0 + 48, kn, DIM_);
    }
    acc[0] = wmma_bf16(a, b0, acc[0]);
    acc[1] = wmma_bf16(a, b1, acc[1]);
    acc[2] = wmma_bf16(a, b2, acc[2]);
    acc[3] = wmma_bf16(a, b3, acc[3]);
    a = an; b0 = c0; b1 = c1; b2 = c2; b3 = c3;
  }
  const int lane = threadIdx.x & 31;
  const int rofs = (lane & 16) ? 8 : 0;
  const int col  = lane & 15;
#pragma unroll
  for (int j = 0; j < 4; ++j)
#pragma unroll
    for (int r = 0; r < 8; ++r)
      out[(size_t)(row0 + r + rofs) * DIM_ + col0 + j * 16 + col] = acc[j][r];
}

// ---------------------------------------------------------------------------
// Host launcher
// ---------------------------------------------------------------------------
extern "C" void kernel_launch(void* const* d_in, const int* in_sizes, int n_in,
                              void* d_out, int out_size, void* d_ws,
                              size_t ws_size, hipStream_t stream) {
  const float*     hs   = (const float*)d_in[0];
  const float*     wqkv = (const float*)d_in[1];
  const float*     wo   = (const float*)d_in[2];
  const int*       cu   = (const int*)d_in[3];
  const long long* idx  = (const long long*)d_in[4];
  float* out = (float*)d_out;

  char* ws = (char*)d_ws;
  bf16_t* hsb   = (bf16_t*)(ws);                     // 10,485,760 B
  bf16_t* wqb   = (bf16_t*)(ws + 10485760);          //  6,291,456 B
  bf16_t* wob   = (bf16_t*)(ws + 16777216);          //  2,097,152 B
  bf16_t* qkvb  = (bf16_t*)(ws + 18874368);          // 31,457,280 B
  bf16_t* Qp    = (bf16_t*)(ws + 50331648);          // 16,777,216 B
  bf16_t* Kp    = (bf16_t*)(ws + 67108864);          // 16,777,216 B
  bf16_t* Vt    = (bf16_t*)(ws + 83886080);          // 16,777,216 B
  bf16_t* attnb = (bf16_t*)(ws + 100663296);         // 10,485,760 B

  cvt_kernel<<<(TOTAL_ * DIM_ / 8 + 255) / 256, 256, 0, stream>>>(
      hs, hsb, TOTAL_ * DIM_ / 8);
  cvt_kernel<<<(3 * DIM_ * DIM_ / 8 + 255) / 256, 256, 0, stream>>>(
      wqkv, wqb, 3 * DIM_ * DIM_ / 8);
  cvt_kernel<<<(DIM_ * DIM_ / 8 + 255) / 256, 256, 0, stream>>>(
      wo, wob, DIM_ * DIM_ / 8);

  // QKV projection: 320 * 48 wave-tiles, 8 waves/block
  qkv_gemm<<<1920, 256, 0, stream>>>(hsb, wqb, qkvb);

  // RoPE + pad/scatter
  rope_scatter<<<TOTAL_, 256, 0, stream>>>(qkvb, idx, Qp, Kp, Vt);

  // Attention: sum(L/64)*H = 1280 blocks of 4 waves
  attn_kernel<<<1280, 128, 0, stream>>>(Qp, Kp, Vt, cu, attnb);

  // Output projection: 320 * 16 wave-tiles, 8 waves/block
  out_gemm<<<640, 256, 0, stream>>>(attnb, wob, out);
}
